// SDRB_26912265076990
// MI455X (gfx1250) — matrix operand order repaired and checked by reading, not compiled, not run on hardware
//
#include <hip/hip_runtime.h>
#include <stdint.h>

typedef __attribute__((ext_vector_type(16))) __bf16          bf16x16;
typedef __attribute__((ext_vector_type(16))) unsigned short  u16x16;
typedef __attribute__((ext_vector_type(8)))  unsigned short  u16x8;
typedef __attribute__((ext_vector_type(8)))  float           f32x8;

constexpr int kH = 224, kW = 224, kC = 64, kN = 4;
constexpr int kHW = kH * kW;                    // 50176
constexpr int kStripsW = kW / 32;               // 7 conv strips per row
constexpr int kConvWGs = kN * kH * kStripsW;    // 6272
constexpr int kDWChunks = kW / 56;              // 4 dw chunks per row
constexpr int kDWWGs = kN * kH * kDWChunks;     // 3584
constexpr float kEps = 1e-5f;

// ---------------------------------------------------------------- helpers
__device__ __forceinline__ unsigned short f2bf(float f) {
  __bf16 b = (__bf16)f;
  unsigned short u;
  __builtin_memcpy(&u, &b, 2);
  return u;
}

// CDNA5 async global->LDS copy (ASYNCcnt-tracked), GVS addressing:
// mem = SGPR64 base + VGPR32 offset ; LDS byte address from VGPR (wave-relative)
__device__ __forceinline__ void async_ld_b128(unsigned lds_off,
                                              unsigned long long gbase,
                                              unsigned goff) {
  asm volatile("global_load_async_to_lds_b128 %0, %1, %2"
               :: "v"(lds_off), "v"(goff), "s"(gbase) : "memory");
}
__device__ __forceinline__ void wait_async0() {
  asm volatile("s_wait_asynccnt 0" ::: "memory");
}

// ---------------------------------------------------------------- kpack
// Pack w_d (64,64,3,3) f32 -> bf16 B-fragments: [kslice=18][ntile=4][lane=32][i=16]
// k-slice s: tap = s>>1 (ky*3+kx), ci = (s&1)*32 + kk, kk = (lane>>4)*8 + (i&7) + (i>>3)*16,
// co = ntile*16 + (lane&15)   (matches V_WMMA 32x16 16-bit B layout)
__global__ __launch_bounds__(256) void kpack(const float* __restrict__ wd,
                                             unsigned short* __restrict__ bPack) {
  int e = blockIdx.x * 256 + threadIdx.x;
  if (e >= 18 * 4 * 32 * 16) return;
  int i    = e & 15;
  int lane = (e >> 4) & 31;
  int nt   = (e >> 9) & 3;
  int s    = e >> 11;
  int co   = nt * 16 + (lane & 15);
  int kk   = (lane >> 4) * 8 + (i & 7) + (i >> 3) * 16;
  int tap  = s >> 1;
  int ci   = (s & 1) * 32 + kk;
  bPack[e] = f2bf(wd[(co * 64 + ci) * 9 + tap]);
}

// ---------------------------------------------------------------- kdw
// t = sharedDW3x3(in) + in, written NHWC bf16 (channel-fastest) via LDS transpose.
// STAGE==2: in-values get BN1 affine + relu applied first (in = o1_pre).
template <int STAGE>
__global__ __launch_bounds__(256) void kdw(const float* __restrict__ src,
                                           const float* __restrict__ wsh,
                                           const float* __restrict__ scl,
                                           const float* __restrict__ sft,
                                           unsigned short* __restrict__ tOut) {
  __shared__ unsigned short tile[56 * 66];
  const int wg = blockIdx.x;
  const int n  = wg / (kH * kDWChunks);
  const int rm = wg % (kH * kDWChunks);
  const int h  = rm / kDWChunks;
  const int wBase = (rm % kDWChunks) * 56;
  const int tid = threadIdx.x;

  for (int e = tid; e < 64 * 56; e += 256) {
    int ci = e / 56, wl = e % 56, w = wBase + wl;
    size_t plane = (size_t)(n * kC + ci) * kH;
    float sc = 0.f, sh = 0.f;
    if (STAGE == 2) { sc = scl[ci]; sh = sft[ci]; }
    float acc = 0.f, center = 0.f;
#pragma unroll
    for (int dy = -1; dy <= 1; ++dy) {
#pragma unroll
      for (int dx = -1; dx <= 1; ++dx) {
        int hh = h + dy, ww = w + dx;
        float v = 0.f;
        if ((unsigned)hh < (unsigned)kH && (unsigned)ww < (unsigned)kW) {
          v = src[(plane + hh) * kW + ww];
          if (STAGE == 2) v = fmaxf(fmaf(sc, v, sh), 0.f);
        }
        acc = fmaf(wsh[(dy + 1) * 3 + (dx + 1)], v, acc);
        if (dy == 0 && dx == 0) center = v;
      }
    }
    tile[wl * 66 + ci] = f2bf(acc + center);
  }
  __syncthreads();
  // coalesced NHWC write, 2 channels (one dword) per thread
  unsigned* dst = (unsigned*)(tOut + ((size_t)(n * kH + h) * kW + wBase) * kC);
  for (int e = tid; e < 56 * 32; e += 256) {
    int wl = e >> 5;
    int c2 = (e & 31) * 2;
    unsigned lo = tile[wl * 66 + c2];
    unsigned hi = tile[wl * 66 + c2 + 1];
    dst[wl * 32 + (c2 >> 1)] = lo | (hi << 16);
  }
}

// ---------------------------------------------------------------- kconv
// Implicit-GEMM dense 3x3 conv via v_wmma_f32_16x16x32_bf16.
// WG = 256 thr = 8 waves, tile = 32 px (one row strip) x 64 cout at fixed (n,h).
// wave w: mtile = w&1 (16 px), ntile = w>>1 (16 cout). K = 18 slices
// (slice s: tap = s>>1, ci half = (s&1)*32).
// LDS A-tile: [row 0..2][col 0..33][slot of 9 x 8-half groups, 144B] filled by
// async DMA from the NHWC bf16 t-buffer (stride 144B -> conflict-free banks).
// STAGE==1: residual = x, out <- o1_pre. STAGE==2: residual = relu(bn1(out)),
// out <- o2_pre in place. Per-WG BN partial sums written in fixed order.
template <int STAGE>
__global__ __launch_bounds__(256) void kconv(const unsigned short* __restrict__ tG,
                                             const unsigned short* __restrict__ bPack,
                                             const float* __restrict__ bias,
                                             const float* __restrict__ resid,
                                             const float* __restrict__ scl,
                                             const float* __restrict__ sft,
                                             float* __restrict__ out,
                                             float* __restrict__ partials) {
  __shared__ u16x8 tileA[3 * 34 * 9];     // 14688 B
  __shared__ float outT[32 * 68];         // staged output tile
  __shared__ float stLDS[64 * 4 * 2];     // fixed-order stat combine

  const int wg = blockIdx.x;
  const int n  = wg / (kH * kStripsW);
  const int rm = wg % (kH * kStripsW);
  const int h  = rm / kStripsW;
  const int wBase = (rm % kStripsW) * 32;
  const int tid = threadIdx.x;

  // ---- async fill of A tile: rows h-1..h+1, cols wBase-1..wBase+32, 64 ch
  const unsigned long long tBase =
      (unsigned long long)(uintptr_t)(tG + (size_t)n * kHW * kC);
  for (int c = tid; c < 3 * 34 * 8; c += 256) {
    int r = c / 272, q = c % 272, col = q >> 3, cc = q & 7;
    int gr = h - 1 + r, gw = wBase - 1 + col;
    int idx = (r * 34 + col) * 9 + cc;
    if ((unsigned)gr < (unsigned)kH && (unsigned)gw < (unsigned)kW) {
      unsigned lds  = (unsigned)(uintptr_t)&tileA[idx];
      unsigned goff = (unsigned)(((gr * kW + gw) * kC + cc * 8) * 2);
      async_ld_b128(lds, tBase, goff);
    } else {
      u16x8 z = {0, 0, 0, 0, 0, 0, 0, 0};
      tileA[idx] = z;   // zero padding (conv pad=1)
    }
  }
  wait_async0();
  __syncthreads();

  // ---- WMMA main loop
  const int lane  = tid & 31;
  const int wv    = tid >> 5;
  const int mtile = wv & 1;
  const int ntile = wv >> 1;
  const int mloc  = lane & 15;
  const int hi16  = lane >> 4;
  const int kb8   = hi16 * 8;          // A-frag K-run base within slice
  const int m     = mtile * 16 + mloc; // pixel within strip

  f32x8 acc = {0.f, 0.f, 0.f, 0.f, 0.f, 0.f, 0.f, 0.f};
#pragma unroll
  for (int s = 0; s < 18; ++s) {
    const int tap = s >> 1;
    const int ky = tap / 3, kx = tap - ky * 3;
    const int cib = (s & 1) * 32;
    const int slot = (ky * 34 + (m + kx)) * 9;
    const int g0 = (cib + kb8) >> 3;
    u16x8 lo = tileA[slot + g0];        // K = kb8 .. kb8+7
    u16x8 hi = tileA[slot + g0 + 2];    // K = kb8+16 .. kb8+23
    u16x16 ar = __builtin_shufflevector(lo, hi, 0, 1, 2, 3, 4, 5, 6, 7,
                                        8, 9, 10, 11, 12, 13, 14, 15);
    u16x16 br = *(const u16x16*)(bPack + ((size_t)((s * 4 + ntile) * 32 + lane)) * 16);
    bf16x16 a = __builtin_bit_cast(bf16x16, ar);
    bf16x16 b = __builtin_bit_cast(bf16x16, br);
    acc = __builtin_amdgcn_wmma_f32_16x16x32_bf16(false, a, false, b,
                                                  (short)0, acc, false, false);
  }

  // ---- stage accumulators to LDS (C/D layout: N=lane&15, M=r+8*(lane>>4))
  {
    const int pbase = mtile * 16 + hi16 * 8;
    const int co = ntile * 16 + mloc;
#pragma unroll
    for (int r = 0; r < 8; ++r) outT[(pbase + r) * 68 + co] = acc[r];
  }
  __syncthreads();

  // ---- epilogue: bias + residual, coalesced NCHW store, BN partials
  {
    const int co = tid >> 2;
    const int wseg = (tid & 3) * 8;
    const size_t base = ((size_t)(n * kC + co) * kH + h) * kW + wBase + wseg;
    const float b0 = bias[co];
    float sc = 0.f, sh = 0.f;
    if (STAGE == 2) { sc = scl[co]; sh = sft[co]; }
    float sS = 0.f, sQ = 0.f;
#pragma unroll
    for (int j = 0; j < 8; ++j) {
      float v = outT[(wseg + j) * 68 + co] + b0;
      if (STAGE == 1) {
        v += resid[base + j];                       // + x
      } else {
        float ov = out[base + j];                   // o1_pre (same elem, same thread)
        v += fmaxf(fmaf(sc, ov, sh), 0.f);          // + relu(bn1(o1_pre))
      }
      out[base + j] = v;
      sS += v;
      sQ += v * v;
    }
    stLDS[(co * 4 + (tid & 3)) * 2 + 0] = sS;
    stLDS[(co * 4 + (tid & 3)) * 2 + 1] = sQ;
  }
  __syncthreads();
  if (tid < 64) {
    float S = 0.f, Q = 0.f;
#pragma unroll
    for (int j = 0; j < 4; ++j) {
      S += stLDS[(tid * 4 + j) * 2 + 0];
      Q += stLDS[(tid * 4 + j) * 2 + 1];
    }
    partials[(size_t)wg * 128 + tid] = S;
    partials[(size_t)wg * 128 + 64 + tid] = Q;
  }
}

// ---------------------------------------------------------------- kstats
// One block per channel: deterministic fixed-order reduction of per-WG partials,
// then scale = gamma*rsqrt(var+eps), shift = beta - mean*scale.
__global__ __launch_bounds__(256) void kstats(const float* __restrict__ partials,
                                              const float* __restrict__ gamma,
                                              const float* __restrict__ beta,
                                              float* __restrict__ scl,
                                              float* __restrict__ sft) {
  __shared__ float rs[256], rq[256];
  const int c = blockIdx.x, t = threadIdx.x;
  float S = 0.f, Q = 0.f;
  for (int w = t; w < kConvWGs; w += 256) {
    S += partials[(size_t)w * 128 + c];
    Q += partials[(size_t)w * 128 + 64 + c];
  }
  rs[t] = S; rq[t] = Q;
  __syncthreads();
  for (int st = 128; st > 0; st >>= 1) {
    if (t < st) { rs[t] += rs[t + st]; rq[t] += rq[t + st]; }
    __syncthreads();
  }
  if (t == 0) {
    const float inv = 1.0f / (float)(kN * kHW);
    float mean = rs[0] * inv;
    float var = rq[0] * inv - mean * mean;
    float sc = gamma[c] * rsqrtf(var + kEps);
    scl[c] = sc;
    sft[c] = beta[c] - mean * sc;
  }
}

// ---------------------------------------------------------------- kfinal
// out = relu(bn2(o2_pre)) + x, in place, float4 granularity.
__global__ __launch_bounds__(256) void kfinal(const float* __restrict__ x,
                                              const float* __restrict__ scl,
                                              const float* __restrict__ sft,
                                              float* __restrict__ out) {
  size_t i4 = ((size_t)blockIdx.x * 256 + threadIdx.x) * 4;
  if (i4 >= (size_t)kN * kC * kHW) return;
  int c = (int)((i4 / kHW) & 63);
  float sc = scl[c], sh = sft[c];
  float4 o = *(float4*)(out + i4);
  const float4 xv = *(const float4*)(x + i4);
  o.x = fmaxf(fmaf(sc, o.x, sh), 0.f) + xv.x;
  o.y = fmaxf(fmaf(sc, o.y, sh), 0.f) + xv.y;
  o.z = fmaxf(fmaf(sc, o.z, sh), 0.f) + xv.z;
  o.w = fmaxf(fmaf(sc, o.w, sh), 0.f) + xv.w;
  *(float4*)(out + i4) = o;
}

// ---------------------------------------------------------------- launch
extern "C" void kernel_launch(void* const* d_in, const int* in_sizes, int n_in,
                              void* d_out, int out_size, void* d_ws, size_t ws_size,
                              hipStream_t stream) {
  (void)in_sizes; (void)n_in; (void)out_size; (void)ws_size;
  const float* x    = (const float*)d_in[0];
  const float* wsh1 = (const float*)d_in[1];
  const float* wd1  = (const float*)d_in[2];
  const float* bd1  = (const float*)d_in[3];
  const float* wsh2 = (const float*)d_in[4];
  const float* wd2  = (const float*)d_in[5];
  const float* bd2  = (const float*)d_in[6];
  const float* g1   = (const float*)d_in[7];
  const float* be1  = (const float*)d_in[8];
  const float* g2   = (const float*)d_in[9];
  const float* be2  = (const float*)d_in[10];
  float* out = (float*)d_out;

  char* ws = (char*)d_ws;
  // workspace layout (256B aligned offsets), total ~29.05 MB
  unsigned short* bufT  = (unsigned short*)(ws);                 // 25,690,112 B (bf16 NHWC t)
  unsigned short* bufB1 = (unsigned short*)(ws + 25690112);      //     73,728 B
  unsigned short* bufB2 = (unsigned short*)(ws + 25763840);      //     73,728 B
  float* partials       = (float*)(ws + 25837568);               //  3,211,264 B
  float* scl1           = (float*)(ws + 29048832);
  float* sft1           = scl1 + 64;
  float* scl2           = sft1 + 64;
  float* sft2           = scl2 + 64;

  kpack<<<144, 256, 0, stream>>>(wd1, bufB1);
  kpack<<<144, 256, 0, stream>>>(wd2, bufB2);

  // ---- SDC block 1
  kdw<1><<<kDWWGs, 256, 0, stream>>>(x, wsh1, nullptr, nullptr, bufT);
  kconv<1><<<kConvWGs, 256, 0, stream>>>(bufT, bufB1, bd1, x, nullptr, nullptr,
                                         out, partials);                 // out = o1_pre
  kstats<<<64, 256, 0, stream>>>(partials, g1, be1, scl1, sft1);

  // ---- SDC block 2 (BN1+relu folded into loads of o1_pre)
  kdw<2><<<kDWWGs, 256, 0, stream>>>(out, wsh2, scl1, sft1, bufT);
  kconv<2><<<kConvWGs, 256, 0, stream>>>(bufT, bufB2, bd2, nullptr, scl1, sft1,
                                         out, partials);                 // out = o2_pre
  kstats<<<64, 256, 0, stream>>>(partials, g2, be2, scl2, sft2);

  // ---- out = relu(bn2(o2_pre)) + x
  kfinal<<<12544, 256, 0, stream>>>(x, scl2, sft2, out);
}
